// AttentionResonantFusionModel_75892072121069
// MI455X (gfx1250) — compile-verified, matrix operand order checked by reading, not run on hardware
//
#include <hip/hip_runtime.h>
#include <math.h>
#include <stdint.h>

#define B_   8
#define S_   128
#define D_   256
#define H_   8
#define DH_  32
#define NN_  256
#define V_   32000
#define PHI_ 1.61803398874989484820f

typedef __attribute__((ext_vector_type(16))) __bf16 v16bf;
typedef __attribute__((ext_vector_type(8)))  float  v8f;

// ---------------------------------------------------------------------------
// fp32 -> bf16 convert (optionally sum of two fp32 arrays)
// ---------------------------------------------------------------------------
__global__ __launch_bounds__(256) void cvt_kernel(const float* __restrict__ a,
                                                  const float* __restrict__ b,
                                                  __bf16* __restrict__ dst, int n)
{
    int i = blockIdx.x * blockDim.x + threadIdx.x;
    if (i < n) {
        float v = a[i];
        if (b) v += b[i];
        dst[i] = (__bf16)v;
    }
}

// ---------------------------------------------------------------------------
// Sequential Euler recurrence: each thread owns one (b,d) lane; loop over t.
// Writes X[b,s,d] = h_r_n + h_i_n  and XCB (bf16) = concat(h_r_n, h_i_n).
// ---------------------------------------------------------------------------
__global__ __launch_bounds__(256) void recur_kernel(const int* __restrict__ ids,
                                                    const float* __restrict__ emb,
                                                    float* __restrict__ X,
                                                    __bf16* __restrict__ XCB)
{
    int idx = blockIdx.x * blockDim.x + threadIdx.x;   // 0..B*D-1
    int b = idx / D_;
    int d = idx % D_;
    float hr = 0.f, hi = 0.f;
    for (int t = 0; t < S_; ++t) {
        int tok = ids[b * S_ + t];
        const float* e = emb + (size_t)tok * (2 * D_);
        float w  = e[d];
        float bb = e[D_ + d];
        float tp = (float)t * PHI_;
        float wl = 1.f + fabsf(w);
        float sr, cr, si, ci;
        __sincosf(hr / wl + bb + tp, &sr, &cr);
        __sincosf(hi / wl + bb + tp, &si, &ci);
        float hrn = cr * ci - sr * si;
        float hin = cr * si + sr * ci;
        hr = hrn; hi = hin;
        int m = b * S_ + t;
        X[(size_t)m * D_ + d]               = hrn + hin;
        XCB[(size_t)m * (2 * D_) + d]       = (__bf16)hrn;
        XCB[(size_t)m * (2 * D_) + D_ + d]  = (__bf16)hin;
    }
}

// ---------------------------------------------------------------------------
// Keys are t-independent: Kt[b,s,h,0:32]=cos, [32:64]=sin of cache/wl_k + bk
// ---------------------------------------------------------------------------
__global__ __launch_bounds__(256) void keys_kernel(const float* __restrict__ X,
                                                   const float* __restrict__ wk,
                                                   const float* __restrict__ bk,
                                                   float* __restrict__ Kt)
{
    int idx = blockIdx.x * blockDim.x + threadIdx.x;
    int dh = idx & (DH_ - 1);
    int h  = (idx >> 5) & (H_ - 1);
    int s  = (idx >> 8) & (S_ - 1);
    int b  = idx >> 15;
    float ch  = X[((size_t)(b * S_ + s)) * D_ + h * DH_ + dh];
    float wlk = 1.f + fabsf(wk[h * DH_ + dh]);
    float sk, ck;
    __sincosf(ch / wlk + bk[h * DH_ + dh], &sk, &ck);
    float* kp = Kt + ((size_t)((b * S_ + s) * H_ + h)) * (2 * DH_);
    kp[dh]       = ck;
    kp[DH_ + dh] = sk;
}

// ---------------------------------------------------------------------------
// Per (t,b,h): build q (64), masked scores over s, softmax, zero masked wgts.
// ---------------------------------------------------------------------------
__global__ __launch_bounds__(128) void attn_kernel(const float* __restrict__ X,
                                                   const float* __restrict__ wq,
                                                   const float* __restrict__ bq,
                                                   const float* __restrict__ Kt,
                                                   float* __restrict__ wattn)
{
    __shared__ float qsh[2 * DH_];
    __shared__ float buf[S_];
    int t = blockIdx.x;
    int b = blockIdx.y / H_;
    int h = blockIdx.y % H_;
    int tid = threadIdx.x;
    float tp = (float)t * PHI_;
    if (tid < DH_) {
        float xh  = X[((size_t)(b * S_ + t)) * D_ + h * DH_ + tid];
        float wlq = 1.f + fabsf(wq[h * DH_ + tid]);
        float sq, cq;
        __sincosf(xh / wlq + bq[h * DH_ + tid] + tp, &sq, &cq);
        qsh[tid]       = cq;
        qsh[DH_ + tid] = sq;
    }
    __syncthreads();
    int s = tid;
    float score = -1e9f;
    if (s < t) {
        const float* kp = Kt + ((size_t)((b * S_ + s) * H_ + h)) * (2 * DH_);
        float dot = 0.f;
#pragma unroll
        for (int e2 = 0; e2 < 2 * DH_; ++e2) dot += qsh[e2] * kp[e2];
        score = dot * 0.125f;   // / sqrt(2*DH)
    }
    buf[tid] = score;
    __syncthreads();
    for (int o = S_ / 2; o > 0; o >>= 1) {
        if (tid < o) buf[tid] = fmaxf(buf[tid], buf[tid + o]);
        __syncthreads();
    }
    float mx = buf[0];
    __syncthreads();
    float ev = __expf(score - mx);
    buf[tid] = ev;
    __syncthreads();
    for (int o = S_ / 2; o > 0; o >>= 1) {
        if (tid < o) buf[tid] += buf[tid + o];
        __syncthreads();
    }
    float z = buf[0];
    float wgt = (s < t) ? ev / z : 0.f;   // cache[s>=t]==0 in reference
    wattn[((size_t)((b * S_ + t) * H_ + h)) * S_ + s] = wgt;
}

// ---------------------------------------------------------------------------
// ctx[b,t,d] = sum_s (sum_h w_attn[b,t,h,s]) * X[b,s,d]   -> bf16
// ---------------------------------------------------------------------------
__global__ __launch_bounds__(256) void ctx_kernel(const float* __restrict__ wattn,
                                                  const float* __restrict__ X,
                                                  __bf16* __restrict__ ctxB)
{
    __shared__ float wsum[S_];
    int t = blockIdx.x, b = blockIdx.y, tid = threadIdx.x;
    if (tid < S_) {
        float acc = 0.f;
#pragma unroll
        for (int h = 0; h < H_; ++h)
            acc += wattn[((size_t)((b * S_ + t) * H_ + h)) * S_ + tid];
        wsum[tid] = acc;
    }
    __syncthreads();
    int d = tid;
    float acc = 0.f;
    for (int s = 0; s < S_; ++s)
        acc += wsum[s] * X[((size_t)(b * S_ + s)) * D_ + d];
    ctxB[((size_t)(b * S_ + t)) * D_ + d] = (__bf16)acc;
}

// ---------------------------------------------------------------------------
// Resonant reduction: ct_sum/st_sum[m,n] = sum_d cos/sin(xc/wl_res+Bres+tp)
// ---------------------------------------------------------------------------
__global__ __launch_bounds__(256) void reson_kernel(const float* __restrict__ xcF,
                                                    const float* __restrict__ Wres,
                                                    const float* __restrict__ Bres,
                                                    __bf16* __restrict__ ctB,
                                                    __bf16* __restrict__ stB)
{
    __shared__ float xr[D_];
    int t = blockIdx.x, b = blockIdx.y, n = threadIdx.x;
    int m = b * S_ + t;
    xr[n] = xcF[(size_t)m * D_ + n];
    __syncthreads();
    float tp = (float)t * PHI_;
    const float* wr = Wres + (size_t)n * D_;
    const float* br = Bres + (size_t)n * D_;
    float cs = 0.f, ss = 0.f;
    for (int d = 0; d < D_; ++d) {
        float wl = 1.f + fabsf(wr[d]);
        float st, ct;
        __sincosf(xr[d] / wl + br[d] + tp, &st, &ct);
        cs += ct; ss += st;
    }
    ctB[(size_t)m * NN_ + n] = (__bf16)cs;
    stB[(size_t)m * NN_ + n] = (__bf16)ss;
}

// ---------------------------------------------------------------------------
// Generic bf16 WMMA GEMM:  C[M,N] (+)= A[M,K] @ Bm[N,K]^T  (+ bias[n])
// Block = 8 waves, 128(M) x 64(N) tile; wave -> 16x16 x 4 N-tiles.
// B panel (64 rows x 32 k, 4KB) is staged in LDS once per block via the
// CDNA5 async-to-LDS path (global_load_async_to_lds_b128), double-buffered:
// copy of chunk k+32 overlaps ds_load + v_wmma of chunk k. Cuts B global
// traffic 8x (was loaded redundantly by all 8 waves).
// ---------------------------------------------------------------------------
__device__ __forceinline__ v16bf load_frag(const __bf16* p)
{
    union { v16bf v; uint4 q[2]; } u;
    u.q[0] = *reinterpret_cast<const uint4*>(p);        // K = kc+8h   .. +7
    u.q[1] = *reinterpret_cast<const uint4*>(p + 16);   // K = kc+8h+16.. +23
    return u.v;
}

__device__ __forceinline__ void async_cp16(uint32_t lds_addr, const __bf16* gp)
{
    // VDST = LDS byte address (low 32 bits of the flat LDS-aperture address),
    // VADDR = 64-bit global address. Tracked by ASYNCcnt.
    asm volatile("global_load_async_to_lds_b128 %0, %1, off"
                 :: "v"(lds_addr), "v"(gp)
                 : "memory");
}

__device__ __forceinline__ void wait_async()
{
    asm volatile("s_wait_asynccnt 0x0" ::: "memory");
}

__global__ __launch_bounds__(256) void wmma_gemm_bf16(const __bf16* __restrict__ A,
                                                      const __bf16* __restrict__ Bm,
                                                      float* __restrict__ C,
                                                      const float* __restrict__ bias,
                                                      int M, int N, int K, int beta)
{
    __shared__ __align__(16) __bf16 bpanel[2][64 * 32];   // 2 x 4KB panels

    const int tid   = threadIdx.x;
    const int lane  = tid & 31;
    const int wave  = tid >> 5;
    const int half  = lane >> 4;
    const int lr    = lane & 15;
    const int mBase = blockIdx.y * 128 + wave * 16;
    const int nBase = blockIdx.x * 64;

    v8f acc[4] = {v8f{}, v8f{}, v8f{}, v8f{}};

    const __bf16* aRow = A + (size_t)(mBase + lr) * K + 8 * half;

    // async-copy assignment: thread -> (row n = tid/4, 16B segment = tid%4)
    const int cpRow = tid >> 2;          // 0..63
    const int cpSeg = tid & 3;           // 0..3
    const __bf16* gB = Bm + (size_t)(nBase + cpRow) * K + cpSeg * 8;
    // Low 32 bits of a flat LDS-aperture address == LDS byte offset (ISA 10.2)
    const uint32_t lmine =
        (uint32_t)(uintptr_t)(const void*)&bpanel[0][cpRow * 32 + cpSeg * 8];

    // prologue: stage k-chunk 0 into buffer 0
    async_cp16(lmine, gB);
    wait_async();
    __syncthreads();

    int buf = 0;
    for (int kc = 0; kc < K; kc += 32) {
        const bool more = (kc + 32) < K;
        if (more) {
            async_cp16(lmine + (uint32_t)((buf ^ 1) * (64 * 32 * 2)), gB + kc + 32);
            __builtin_prefetch((const void*)(aRow + kc + 32), 0, 1);
        }

        v16bf a = load_frag(aRow + kc);                     // global b128 x2

        const __bf16* lp = &bpanel[buf][(size_t)lr * 32 + 8 * half];
        v16bf f0 = load_frag(lp);                           // ds_load_b128 x2
        v16bf f1 = load_frag(lp + 16 * 32);
        v16bf f2 = load_frag(lp + 32 * 32);
        v16bf f3 = load_frag(lp + 48 * 32);

        acc[0] = __builtin_amdgcn_wmma_f32_16x16x32_bf16(false, a, false, f0,
                                                         (short)0, acc[0], false, false);
        acc[1] = __builtin_amdgcn_wmma_f32_16x16x32_bf16(false, a, false, f1,
                                                         (short)0, acc[1], false, false);
        acc[2] = __builtin_amdgcn_wmma_f32_16x16x32_bf16(false, a, false, f2,
                                                         (short)0, acc[2], false, false);
        acc[3] = __builtin_amdgcn_wmma_f32_16x16x32_bf16(false, a, false, f3,
                                                         (short)0, acc[3], false, false);

        if (more) {
            wait_async();       // copy of next chunk complete (this wave)
            __syncthreads();    // visible to all waves; prior reads done
            buf ^= 1;
        }
    }

#pragma unroll
    for (int j = 0; j < 4; ++j) {
        int n = nBase + j * 16 + lr;
        float bb = bias ? bias[n] : 0.f;
#pragma unroll
        for (int v = 0; v < 8; ++v) {
            size_t idx = (size_t)(mBase + v + 8 * half) * N + n;
            float val = acc[j][v] + bb;
            if (beta) val += C[idx];
            C[idx] = val;
        }
    }
}

// ---------------------------------------------------------------------------
extern "C" void kernel_launch(void* const* d_in, const int* in_sizes, int n_in,
                              void* d_out, int out_size, void* d_ws, size_t ws_size,
                              hipStream_t stream)
{
    const int*   ids  = (const int*)  d_in[0];
    const float* emb  = (const float*)d_in[1];
    const float* wq   = (const float*)d_in[2];
    const float* bq   = (const float*)d_in[3];
    const float* wk   = (const float*)d_in[4];
    const float* bk   = (const float*)d_in[5];
    const float* Wr   = (const float*)d_in[6];
    const float* Wi   = (const float*)d_in[7];
    const float* Wc   = (const float*)d_in[8];
    const float* bc   = (const float*)d_in[9];
    const float* Wres = (const float*)d_in[10];
    const float* Bres = (const float*)d_in[11];
    const float* Wor  = (const float*)d_in[12];
    const float* Woi  = (const float*)d_in[13];
    const float* Wout = (const float*)d_in[14];
    float* out = (float*)d_out;

    char* ws = (char*)d_ws;
    size_t off = 0;
    auto carve = [&](size_t bytes) -> void* {
        void* p = ws + off;
        off += (bytes + 255) & ~(size_t)255;
        return p;
    };

    const int M = B_ * S_;                       // 1024 rows, m = b*S + t
    __bf16* WoutB = (__bf16*)carve((size_t)V_ * D_ * 2);
    __bf16* WcB   = (__bf16*)carve((size_t)D_ * 2 * D_ * 2);
    __bf16* WriB  = (__bf16*)carve((size_t)D_ * D_ * 2);
    __bf16* WorB  = (__bf16*)carve((size_t)D_ * NN_ * 2);
    __bf16* WoiB  = (__bf16*)carve((size_t)D_ * NN_ * 2);
    float*  X     = (float*) carve((size_t)M * D_ * 4);
    __bf16* XCB   = (__bf16*)carve((size_t)M * 2 * D_ * 2);
    float*  Kt    = (float*) carve((size_t)B_ * S_ * H_ * 2 * DH_ * 4);
    float*  watt  = (float*) carve((size_t)M * H_ * S_ * 4);
    __bf16* ctxB  = (__bf16*)carve((size_t)M * D_ * 2);
    float*  xcF   = (float*) carve((size_t)M * D_ * 4);
    __bf16* ctB   = (__bf16*)carve((size_t)M * NN_ * 2);
    __bf16* stB   = (__bf16*)carve((size_t)M * NN_ * 2);
    float*  Ybuf  = (float*) carve((size_t)M * D_ * 4);
    __bf16* YB    = (__bf16*)carve((size_t)M * D_ * 2);

    // Weight conversions to bf16 (Wri = Wr + Wi merged once)
    cvt_kernel<<<(V_ * D_ + 255) / 256, 256, 0, stream>>>(Wout, nullptr, WoutB, V_ * D_);
    cvt_kernel<<<(D_ * 2 * D_ + 255) / 256, 256, 0, stream>>>(Wc, nullptr, WcB, D_ * 2 * D_);
    cvt_kernel<<<(D_ * D_ + 255) / 256, 256, 0, stream>>>(Wr, Wi, WriB, D_ * D_);
    cvt_kernel<<<(D_ * NN_ + 255) / 256, 256, 0, stream>>>(Wor, nullptr, WorB, D_ * NN_);
    cvt_kernel<<<(D_ * NN_ + 255) / 256, 256, 0, stream>>>(Woi, nullptr, WoiB, D_ * NN_);

    // Sequential state recurrence (parallel over B*D lanes)
    recur_kernel<<<(B_ * D_) / 256, 256, 0, stream>>>(ids, emb, X, XCB);

    // Key tensor (t-independent)
    keys_kernel<<<(B_ * S_ * H_ * DH_) / 256, 256, 0, stream>>>(X, wk, bk, Kt);

    // Attention weights, head sum + context
    attn_kernel<<<dim3(S_, B_ * H_), 128, 0, stream>>>(X, wq, bq, Kt, watt);
    ctx_kernel<<<dim3(S_, B_), 256, 0, stream>>>(watt, X, ctxB);

    // xc = concat(h_r,h_i) @ Wc^T + bc      (WMMA, K=512)
    wmma_gemm_bf16<<<dim3(D_ / 64, M / 128), 256, 0, stream>>>(
        XCB, WcB, xcF, bc, M, D_, 2 * D_, 0);

    // Resonant sincos reduction
    reson_kernel<<<dim3(S_, B_), 256, 0, stream>>>(xcF, Wres, Bres, ctB, stB);

    // Y = ctx@(Wr+Wi)^T + ct_sum@Wor^T + st_sum@Woi^T   (WMMA x3)
    wmma_gemm_bf16<<<dim3(D_ / 64, M / 128), 256, 0, stream>>>(
        ctxB, WriB, Ybuf, nullptr, M, D_, D_, 0);
    wmma_gemm_bf16<<<dim3(D_ / 64, M / 128), 256, 0, stream>>>(
        ctB, WorB, Ybuf, nullptr, M, D_, NN_, 1);
    wmma_gemm_bf16<<<dim3(D_ / 64, M / 128), 256, 0, stream>>>(
        stB, WoiB, Ybuf, nullptr, M, D_, NN_, 1);

    cvt_kernel<<<(M * D_ + 255) / 256, 256, 0, stream>>>(Ybuf, nullptr, YB, M * D_);

    // logits = Y @ Wout^T   (the dominant GEMM: 1024 x 32000 x 256, WMMA)
    wmma_gemm_bf16<<<dim3(V_ / 64, M / 128), 256, 0, stream>>>(
        YB, WoutB, out, nullptr, M, V_, D_, 0);
}
